// TranceptionAttention_62045097558328
// MI455X (gfx1250) — matrix operand order repaired and checked by reading, not compiled, not run on hardware
//
#include <hip/hip_runtime.h>
#include <hip/hip_bf16.h>

// ---------------------------------------------------------------------------
// Live computation (after dead-code elimination of the reference):
//   Q  = x @ Wq[256:1024].T            (8192 x 768)   -> ws
//   VF[:, 0:256]   = x @ Wv[0:256].T                  -> ws
//   VF[:, 256:1024] = dwconv(wv, dwconv(wq, Q))  per group (ks=3,5,7)
//   out = VF @ Wo.T + bo               (8192 x 1024)
// GEMMs use bf16 WMMA with a hi/lo split (3 products) for ~fp32 accuracy.
// alibi / Wk / k-convs / softmax / attention einsums are dead code.
// ---------------------------------------------------------------------------

typedef __attribute__((ext_vector_type(16))) __bf16 v16bf;
typedef __attribute__((ext_vector_type(8)))  __bf16 v8bf;
typedef __attribute__((ext_vector_type(8)))  float  v8f;

__device__ __forceinline__ unsigned short f2bf(float f) {
  unsigned u = __float_as_uint(f);
  u += 0x7FFFu + ((u >> 16) & 1u);          // round-to-nearest-even
  return (unsigned short)(u >> 16);
}
__device__ __forceinline__ float bf2f(unsigned short h) {
  return __uint_as_float(((unsigned)h) << 16);
}

__device__ __forceinline__ v16bf ldfrag(const unsigned short* p) {
  v8bf c0 = *(const v8bf*)p;                 // 16B-aligned ds_load_b128
  v8bf c1 = *(const v8bf*)(p + 8);
  return __builtin_shufflevector(c0, c1, 0,1,2,3,4,5,6,7,8,9,10,11,12,13,14,15);
}

#define MBLK 128
#define NBLK 64
#define KBLK 32
#define LDS_STRIDE 40   // 32 elems + 8 pad (shorts); 16B-aligned chunks, conflict-free

// C[m, n] = sum_k A[m, k] * W[n + row_off, k]   (+ bias[n] if HAS_BIAS)
// Grid: x = N/64, y = M/128. Block: 256 threads (8 waves).
template <bool HAS_BIAS>
__global__ __launch_bounds__(256)
void gemm_bf16x3(const float* __restrict__ A, int lda,
                 const float* __restrict__ W, int ldw, int row_off,
                 const float* __restrict__ bias,
                 float* __restrict__ C, int ldc, int K)
{
  __shared__ unsigned short Ah[MBLK * LDS_STRIDE];
  __shared__ unsigned short Al[MBLK * LDS_STRIDE];
  __shared__ unsigned short Bh[NBLK * LDS_STRIDE];
  __shared__ unsigned short Bl[NBLK * LDS_STRIDE];

  const int tid  = threadIdx.x;
  const int lane = tid & 31;
  const int wave = tid >> 5;
  const int lm = lane & 15, lh = lane >> 4;
  const int wm = wave >> 1;                 // 0..3 : wave row (32 rows each)
  const int wn = wave & 1;                  // 0..1 : wave col (32 cols each)

  const int m0 = blockIdx.y * MBLK;
  const int n0 = blockIdx.x * NBLK;

  // Global->LDS assignments
  const int ar = tid >> 1, ahalf = tid & 1; // A: 128 rows x 2 halves of 16 floats
  const int bn = tid >> 2, bq    = tid & 3; // B: 64 rows x 4 quarters of 8 floats

  const float* Agp = A + (size_t)(m0 + ar) * lda + 16 * ahalf;
  const float* Bgp = W + (size_t)(n0 + bn + row_off) * ldw + 8 * bq;

  // A stored permuted per the ISA 16-bit A-fragment layout:
  //   row order [k0..7, k16..23, k8..15, k24..31] -> each lane's fragment is
  //   16 contiguous bf16 at offset 16*(lane/16).
  unsigned short* ApH = &Ah[ar * LDS_STRIDE + 8 * ahalf];
  unsigned short* ApL = &Al[ar * LDS_STRIDE + 8 * ahalf];
  unsigned short* BpH = &Bh[bn * LDS_STRIDE + 8 * (2 * (bq & 1) + (bq >> 1))];
  unsigned short* BpL = &Bl[bn * LDS_STRIDE + 8 * (2 * (bq & 1) + (bq >> 1))];

  v8f acc[2][2] = {};

  // ---- register double-buffer: prefetch tile 0 ----
  float a[16], b[8];
  #pragma unroll
  for (int i = 0; i < 4; ++i) {
    float4 t = *(const float4*)(Agp + 4 * i);
    a[4*i+0]=t.x; a[4*i+1]=t.y; a[4*i+2]=t.z; a[4*i+3]=t.w;
  }
  #pragma unroll
  for (int i = 0; i < 2; ++i) {
    float4 t = *(const float4*)(Bgp + 4 * i);
    b[4*i+0]=t.x; b[4*i+1]=t.y; b[4*i+2]=t.z; b[4*i+3]=t.w;
  }

  for (int kk = 0; kk < K; kk += KBLK) {
    // ---- convert current tile (fp32 -> bf16 hi/lo) and stage to LDS ----
    #pragma unroll
    for (int i = 0; i < 8; ++i) {
      unsigned short h0 = f2bf(a[i]);
      unsigned short h1 = f2bf(a[8 + i]);
      ApH[i]      = h0;
      ApH[16 + i] = h1;
      ApL[i]      = f2bf(a[i]     - bf2f(h0));
      ApL[16 + i] = f2bf(a[8 + i] - bf2f(h1));
    }
    #pragma unroll
    for (int i = 0; i < 8; ++i) {
      unsigned short h = f2bf(b[i]);
      BpH[i] = h;
      BpL[i] = f2bf(b[i] - bf2f(h));
    }
    __syncthreads();

    // ---- prefetch next tile's globals; overlaps with WMMA below ----
    if (kk + KBLK < K) {
      #pragma unroll
      for (int i = 0; i < 4; ++i) {
        float4 t = *(const float4*)(Agp + kk + KBLK + 4 * i);
        a[4*i+0]=t.x; a[4*i+1]=t.y; a[4*i+2]=t.z; a[4*i+3]=t.w;
      }
      #pragma unroll
      for (int i = 0; i < 2; ++i) {
        float4 t = *(const float4*)(Bgp + kk + KBLK + 4 * i);
        b[4*i+0]=t.x; b[4*i+1]=t.y; b[4*i+2]=t.z; b[4*i+3]=t.w;
      }
    }

    // ---- WMMA compute: 2x2 tiles x 3 split-products ----
    v16bf afh[2], afl[2], bfh[2], bfl[2];
    #pragma unroll
    for (int im = 0; im < 2; ++im) {
      int row = wm * 32 + im * 16 + lm;
      afh[im] = ldfrag(&Ah[row * LDS_STRIDE + 16 * lh]);
      afl[im] = ldfrag(&Al[row * LDS_STRIDE + 16 * lh]);
    }
    #pragma unroll
    for (int in = 0; in < 2; ++in) {
      int col = wn * 32 + in * 16 + lm;
      bfh[in] = ldfrag(&Bh[col * LDS_STRIDE + 16 * lh]);
      bfl[in] = ldfrag(&Bl[col * LDS_STRIDE + 16 * lh]);
    }
    #pragma unroll
    for (int im = 0; im < 2; ++im) {
      #pragma unroll
      for (int in = 0; in < 2; ++in) {
        acc[im][in] = __builtin_amdgcn_wmma_f32_16x16x32_bf16(
            false, afh[im], false, bfh[in], (short)0, acc[im][in], false, false);
        acc[im][in] = __builtin_amdgcn_wmma_f32_16x16x32_bf16(
            false, afl[im], false, bfh[in], (short)0, acc[im][in], false, false);
        acc[im][in] = __builtin_amdgcn_wmma_f32_16x16x32_bf16(
            false, afh[im], false, bfl[in], (short)0, acc[im][in], false, false);
      }
    }
    __syncthreads();
  }

  // ---- epilogue: C layout n = lane%16, m = r + 8*(lane/16); pointer-stepped ----
  #pragma unroll
  for (int im = 0; im < 2; ++im) {
    #pragma unroll
    for (int in = 0; in < 2; ++in) {
      int n = n0 + wn * 32 + in * 16 + lm;
      float bval = HAS_BIAS ? bias[n] : 0.0f;
      float* Cp = C + (size_t)(m0 + wm * 32 + im * 16 + 8 * lh) * ldc + n;
      #pragma unroll
      for (int r = 0; r < 8; ++r) {
        *Cp = acc[im][in][r] + bval;
        Cp += ldc;
      }
    }
  }
}

// ---------------------------------------------------------------------------
// Depthwise causal conv chain: u = conv(wq, q)+bq ; v = conv(wv, u)+bv
// Q: (8192 x 768), col (h'*64 + d) = q head (4+h').  VF cols 256..1023 written.
// Grid: (L/64, 4 heads-in-group, B). One group (ks) per launch.
// ---------------------------------------------------------------------------
#define LT 64
#define HQ 12   // q halo = 2*(ks_max-1)
#define HU 6    // u halo = ks_max-1

__global__ __launch_bounds__(256)
void dwconv_chain(const float* __restrict__ Q, float* __restrict__ VF,
                  const float* __restrict__ wq, const float* __restrict__ bq,
                  const float* __restrict__ wv, const float* __restrict__ bv,
                  int ks, int group)
{
  __shared__ float qt[(LT + HQ) * 64];
  __shared__ float ut[(LT + HU) * 64];

  const int tid = threadIdx.x;
  const int d   = tid & 63;
  const int b   = blockIdx.z;
  const int hh  = group * 4 + blockIdx.y;       // 0..11 (head-4)
  const int l0  = blockIdx.x * LT;

  const float* qbase = Q + (size_t)b * 2048 * 768 + hh * 64;

  float wqr[7], wvr[7];
  for (int j = 0; j < ks; ++j) { wqr[j] = wq[d * ks + j]; wvr[j] = wv[d * ks + j]; }
  const float bqr = bq[d], bvr = bv[d];

  // load q tile rows [l0-HQ, l0+LT)
  for (int idx = tid; idx < (LT + HQ) * 64; idx += 256) {
    int r = idx >> 6, dd = idx & 63;
    int l = l0 - HQ + r;
    qt[idx] = (l < 0) ? 0.0f : qbase[(size_t)l * 768 + dd];
  }
  __syncthreads();

  // u rows [l0-HU, l0+LT)  (u padding: u(l<0) = 0, per reference pad-then-conv)
  for (int idx = tid; idx < (LT + HU) * 64; idx += 256) {
    int r = idx >> 6;                            // idx & 63 == d (256 % 64 == 0)
    int l = l0 - HU + r;
    float u = 0.0f;
    if (l >= 0) {
      u = bqr;
      int qr = r + HU - (ks - 1);
      for (int j = 0; j < ks; ++j) u += wqr[j] * qt[(qr + j) * 64 + d];
    }
    ut[idx] = u;
  }
  __syncthreads();

  // v rows [l0, l0+LT)
  float* vbase = VF + (size_t)b * 2048 * 1024 + 256 + hh * 64;
  for (int idx = tid; idx < LT * 64; idx += 256) {
    int r = idx >> 6;
    int l = l0 + r;
    float v = bvr;
    int ur = r + HU - (ks - 1);
    for (int j = 0; j < ks; ++j) v += wvr[j] * ut[(ur + j) * 64 + d];
    vbase[(size_t)l * 1024 + d] = v;
  }
}

// ---------------------------------------------------------------------------
extern "C" void kernel_launch(void* const* d_in, const int* in_sizes, int n_in,
                              void* d_out, int out_size, void* d_ws, size_t ws_size,
                              hipStream_t stream)
{
  const float* x   = (const float*)d_in[0];
  // d_in[1] = alibi  : dead
  const float* Wq  = (const float*)d_in[2];
  // d_in[3] = Wk     : dead
  const float* Wv  = (const float*)d_in[4];
  const float* Wo  = (const float*)d_in[5];
  const float* bo  = (const float*)d_in[6];
  const float* wq3 = (const float*)d_in[7];  const float* bq3 = (const float*)d_in[8];
  const float* wv3 = (const float*)d_in[11]; const float* bv3 = (const float*)d_in[12];
  const float* wq5 = (const float*)d_in[13]; const float* bq5 = (const float*)d_in[14];
  const float* wv5 = (const float*)d_in[17]; const float* bv5 = (const float*)d_in[18];
  const float* wq7 = (const float*)d_in[19]; const float* bq7 = (const float*)d_in[20];
  const float* wv7 = (const float*)d_in[23]; const float* bv7 = (const float*)d_in[24];
  float* out = (float*)d_out;

  const int M = 4 * 2048;   // 8192 rows
  const int K = 1024;

  float* Qws = (float*)d_ws;                 // 8192 x 768
  float* VF  = Qws + (size_t)M * 768;        // 8192 x 1024  (total ws ~59 MB)

  dim3 blk(256);

  // Stage A1: Q (heads 4..15) = x @ Wq[256:1024].T
  gemm_bf16x3<false><<<dim3(768 / NBLK, M / MBLK), blk, 0, stream>>>(
      x, 1024, Wq, 1024, /*row_off=*/256, nullptr, Qws, 768, K);

  // Stage A2: VF[:, 0:256] (heads 0..3) = x @ Wv[0:256].T
  gemm_bf16x3<false><<<dim3(256 / NBLK, M / MBLK), blk, 0, stream>>>(
      x, 1024, Wv, 1024, /*row_off=*/0, nullptr, VF, 1024, K);

  // Stage B: depthwise causal conv chains per group
  dim3 cg(2048 / LT, 4, 4);
  dwconv_chain<<<cg, blk, 0, stream>>>(Qws, VF, wq3, bq3, wv3, bv3, 3, 0);
  dwconv_chain<<<cg, blk, 0, stream>>>(Qws, VF, wq5, bq5, wv5, bv5, 5, 1);
  dwconv_chain<<<cg, blk, 0, stream>>>(Qws, VF, wq7, bq7, wv7, bv7, 7, 2);

  // Stage C: out = VF @ Wo.T + bo
  gemm_bf16x3<true><<<dim3(1024 / NBLK, M / MBLK), blk, 0, stream>>>(
      VF, 1024, Wo, 1024, /*row_off=*/0, bo, out, 1024, K);
}
